// CausalSelfAttention_37177236914967
// MI455X (gfx1250) — matrix operand order repaired and checked by reading, not compiled, not run on hardware
//
#include <hip/hip_runtime.h>
#include <hip/hip_bf16.h>

// CDNA5 / gfx1250 causal self-attention:
//   qkv = x @ W_qkv + b_qkv   (bf16 WMMA GEMM, fp32 acc, bf16 out, async-LDS double buffer)
//   flash-attention per (b,h)  (bf16 WMMA, online softmax, fp32 acc, async K staging)
//   out = y @ W_proj + b_proj (bf16 WMMA GEMM, fp32 acc, fp32 out)

typedef __bf16 bf16;
typedef __attribute__((ext_vector_type(16))) __bf16 v16bf;
typedef __attribute__((ext_vector_type(8)))  float  v8f;
typedef __attribute__((ext_vector_type(8)))  unsigned int v8u;

__device__ __forceinline__ v8f wmma_bf16(v16bf a, v16bf b, v8f c) {
  return __builtin_amdgcn_wmma_f32_16x16x32_bf16(
      /*neg_a=*/false, a, /*neg_b=*/false, b,
      /*c_mod=*/(short)0, c, /*reuse_a=*/false, /*reuse_b=*/false);
}

// Async DMA-style copy: 16 bytes global -> LDS, tracked by ASYNCcnt.
__device__ __forceinline__ void async_load_b128(unsigned lds_off, const void* gaddr) {
  asm volatile("global_load_async_to_lds_b128 %0, %1, off"
               :: "v"(lds_off), "v"((unsigned long long)gaddr)
               : "memory");
}
__device__ __forceinline__ void wait_async0() {
  asm volatile("s_wait_asynccnt 0" ::: "memory");
}
__device__ __forceinline__ void wait_ds0() {
  asm volatile("s_wait_dscnt 0" ::: "memory");
}
__device__ __forceinline__ unsigned lds_off_of(const void* p) {
  return (unsigned)(unsigned long long)p;  // low 32 bits of generic ptr = LDS offset
}

// A-matrix 16x32 bf16 fragment from LDS (row-major [m][k], k pairs contiguous).
// ISA 7.12.2: lane 0-15 -> M=lane, K pairs: vgpr0-3 = K 0..7 (+8 if lane>=16),
// vgpr4-7 = K 16..23 (+8 if lane>=16).
__device__ __forceinline__ v16bf load_a_frag(const bf16* base, int stride,
                                             int lane, int kbase) {
  const int m = lane & 15;
  const int khalf = (lane & 16) ? 8 : 0;
  v8u u;
#pragma unroll
  for (int p = 0; p < 8; ++p) {
    const int k = kbase + ((p < 4) ? 0 : 16) + khalf + 2 * (p & 3);
    u[p] = *(const unsigned int*)(base + m * stride + k);
  }
  return __builtin_bit_cast(v16bf, u);
}

// B-matrix 32x16 bf16 fragment where LDS holds B transposed: base[n*stride+k]
// (contraction-dim pairs contiguous -> dword/b128 loads).
__device__ __forceinline__ v16bf load_bT_frag(const bf16* base, int stride,
                                              int lane, int nloc, int kbase) {
  const int n = (lane & 15) + nloc;
  const int koff = (lane & 16) ? 8 : 0;
  v8u u;
#pragma unroll
  for (int j = 0; j < 8; ++j) {
    const int k = kbase + ((j < 4) ? 0 : 16) + koff + 2 * (j & 3);
    u[j] = *(const unsigned int*)(base + n * stride + k);
  }
  return __builtin_bit_cast(v16bf, u);
}

// B-matrix fragment where LDS holds B row-major [k][n] (pairs strided).
__device__ __forceinline__ v16bf load_b_frag_strided(const bf16* base, int stride,
                                                     int lane, int nloc) {
  const int n = (lane & 15) + nloc;
  const int koff = (lane & 16) ? 8 : 0;
  v8u u;
#pragma unroll
  for (int j = 0; j < 8; ++j) {
    const int k = ((j < 4) ? 0 : 16) + koff + 2 * (j & 3);
    unsigned short lo = __builtin_bit_cast(unsigned short, base[k * stride + n]);
    unsigned short hi = __builtin_bit_cast(unsigned short, base[(k + 1) * stride + n]);
    u[j] = ((unsigned int)hi << 16) | (unsigned int)lo;
  }
  return __builtin_bit_cast(v16bf, u);
}

// ---------------------------------------------------------------------------
// fp32 -> bf16 conversion
// ---------------------------------------------------------------------------
__global__ void cvt_f32_bf16_kernel(const float* __restrict__ in,
                                    bf16* __restrict__ out, int n) {
  int i = (blockIdx.x * blockDim.x + threadIdx.x) * 4;
  if (i + 3 < n) {
    float4 v = *(const float4*)(in + i);
    out[i + 0] = (bf16)v.x;
    out[i + 1] = (bf16)v.y;
    out[i + 2] = (bf16)v.z;
    out[i + 3] = (bf16)v.w;
  } else {
    for (; i < n; ++i) out[i] = (bf16)in[i];
  }
}

// ---------------------------------------------------------------------------
// Tiled bf16 WMMA GEMM: C[M,N] = A[M,K] @ B[K,N] + bias[N]
// Block tile 128x64, 8 waves (4x2), each wave 32x32 (2x2 WMMA tiles), K-step 32.
// Double-buffered LDS tiles filled with global_load_async_to_lds_b128.
// ---------------------------------------------------------------------------
template <bool OUT_BF16>
__global__ __launch_bounds__(256) void gemm_bf16_kernel(
    const bf16* __restrict__ A, const bf16* __restrict__ B,
    const float* __restrict__ bias, void* __restrict__ Cout,
    int M, int N, int K) {
  __shared__ uint4 sA4[2][128 * 4];  // 128 rows x 32 bf16 (row-major), x2 buffers
  __shared__ uint4 sB4[2][32 * 8];   // 32 rows x 64 bf16 (row-major [k][n]), x2

  const int tid = threadIdx.x;
  const int lane = tid & 31;
  const int wave = tid >> 5;
  const int wr = wave >> 1;   // 0..3
  const int wc = wave & 1;    // 0..1
  const int m0 = blockIdx.y * 128;
  const int n0 = blockIdx.x * 64;

  const int arow = tid >> 1, ahalf = tid & 1;       // A: 256 thr x 32B
  const int brow = tid >> 3, bch = tid & 7;         // B: 256 thr x 16B

  auto stage = [&](int buf, int kk) {
    const bf16* asrc = A + (size_t)(m0 + arow) * K + kk + ahalf * 16;
    unsigned adst = lds_off_of(&sA4[buf][arow * 4 + ahalf * 2]);
    async_load_b128(adst, asrc);
    async_load_b128(adst + 16, asrc + 8);
    const bf16* bsrc = B + (size_t)(kk + brow) * N + n0 + bch * 8;
    async_load_b128(lds_off_of(&sB4[buf][brow * 8 + bch]), bsrc);
  };

  v8f acc[2][2] = {};

  stage(0, 0);
  int cur = 0;
  for (int kk = 0; kk < K; kk += 32) {
    wait_async0();
    __syncthreads();
    if (kk + 32 < K) stage(cur ^ 1, kk + 32);

    const bf16* sA = (const bf16*)sA4[cur];
    const bf16* sB = (const bf16*)sB4[cur];
    v16bf a0 = load_a_frag(sA + (wr * 32 + 0) * 32, 32, lane, 0);
    v16bf a1 = load_a_frag(sA + (wr * 32 + 16) * 32, 32, lane, 0);
    v16bf b0 = load_b_frag_strided(sB, 64, lane, wc * 32 + 0);
    v16bf b1 = load_b_frag_strided(sB, 64, lane, wc * 32 + 16);
    acc[0][0] = wmma_bf16(a0, b0, acc[0][0]);
    acc[0][1] = wmma_bf16(a0, b1, acc[0][1]);
    acc[1][0] = wmma_bf16(a1, b0, acc[1][0]);
    acc[1][1] = wmma_bf16(a1, b1, acc[1][1]);
    cur ^= 1;
  }

  const int hi = lane >> 4, nlo = lane & 15;
#pragma unroll
  for (int mt = 0; mt < 2; ++mt) {
#pragma unroll
    for (int nt = 0; nt < 2; ++nt) {
      const int n = n0 + wc * 32 + nt * 16 + nlo;
      const float bv = bias[n];
#pragma unroll
      for (int r = 0; r < 8; ++r) {
        const int m = m0 + wr * 32 + mt * 16 + r + hi * 8;
        const float v = acc[mt][nt][r] + bv;
        if (OUT_BF16)
          ((bf16*)Cout)[(size_t)m * N + n] = (bf16)v;
        else
          ((float*)Cout)[(size_t)m * N + n] = v;
      }
    }
  }
}

// ---------------------------------------------------------------------------
// Flash attention, causal. One block = (b, h, 128 query rows), 8 waves,
// each wave owns 16 query rows. Key loop in blocks of 64 (causally trimmed).
// D = 64. qkv is bf16 (B,T,3C) packed [q | k | v]; y out bf16 (B,T,C).
// K tile staged via async-to-LDS; V staged transposed for contiguous B-frags.
// ---------------------------------------------------------------------------
__global__ __launch_bounds__(256) void flash_attn_kernel(
    const bf16* __restrict__ qkv, bf16* __restrict__ Y,
    int Bn, int T, int C, int H) {
  constexpr int D = 64;
  __shared__ uint4 sK4[64 * 8];       // K tile [key][d]  64x64 bf16
  __shared__ bf16 sVt[64 * 64];       // V^T   [d][key]   64x64 bf16
  __shared__ bf16 sP[8][16 * 64];     // per-wave P tile [row][key]
  bf16* sK = (bf16*)sK4;

  const int tid = threadIdx.x, lane = tid & 31, wave = tid >> 5;
  const int qblk = blockIdx.x, h = blockIdx.y, b = blockIdx.z;
  const int q0 = qblk * 128;
  const int C3 = 3 * C;
  const float scale = 0.125f;            // 1/sqrt(64)
  const float LOG2E = 1.44269504f;

  const int m = lane & 15;
  const int hi = lane >> 4;
  const int khalf = hi * 8;
  const int nlo = lane & 15;

  // Q A-fragments in registers (16 rows x 64 d = 2 fragments).
  const int trow = q0 + wave * 16 + m;
  const bf16* qrow = qkv + (size_t)(b * T + trow) * C3 + h * D;
  v16bf qf[2];
#pragma unroll
  for (int f = 0; f < 2; ++f) {
    v8u u;
#pragma unroll
    for (int p = 0; p < 8; ++p) {
      const int d = f * 32 + ((p < 4) ? 0 : 16) + khalf + 2 * (p & 3);
      u[p] = *(const unsigned int*)(qrow + d);
    }
    qf[f] = __builtin_bit_cast(v16bf, u);
  }

  float mrow[8], lrow[8];
  v8f Oacc[4] = {};
#pragma unroll
  for (int r = 0; r < 8; ++r) { mrow[r] = -3.0e38f; lrow[r] = 0.0f; }

  const int rowbase = q0 + wave * 16 + hi * 8;
  const int nkblk = 2 * qblk + 2;  // causal: keys <= q0+127

  for (int jb = 0; jb < nkblk; ++jb) {
    const int k0 = jb * 64;
    {  // stage K tile asynchronously; V tile transposed via regular ds stores
      const int kr = tid >> 2, ch = tid & 3;  // 64 rows x 4 chunks (16 bf16)
      const size_t rowoff = (size_t)(b * T + k0 + kr) * C3 + h * D + ch * 16;
      const bf16* ksrc = qkv + rowoff + C;
      unsigned kdst = lds_off_of(&sK4[kr * 8 + ch * 2]);
      async_load_b128(kdst, ksrc);
      async_load_b128(kdst + 16, ksrc + 8);
      const bf16* vsrc = qkv + rowoff + 2 * C;
#pragma unroll
      for (int e = 0; e < 16; ++e)
        sVt[(ch * 16 + e) * 64 + kr] = vsrc[e];
    }
    wait_async0();
    __syncthreads();

    // S = (Q @ K^T) * scale, causal-masked.  B[k=d][n=key] = sK[key*64 + d].
    v8f S[4];
#pragma unroll
    for (int nt = 0; nt < 4; ++nt) {
      v8f s = {};
#pragma unroll
      for (int ks = 0; ks < 2; ++ks) {
        v16bf bfrag = load_bT_frag(sK, 64, lane, nt * 16, ks * 32);
        s = wmma_bf16(qf[ks], bfrag, s);
      }
      S[nt] = s;
    }
#pragma unroll
    for (int nt = 0; nt < 4; ++nt) {
      const int col = k0 + nt * 16 + nlo;
#pragma unroll
      for (int r = 0; r < 8; ++r) {
        const int row = rowbase + r;
        const float sv = S[nt][r] * scale;
        S[nt][r] = (col > row) ? -3.0e38f : sv;
      }
    }

    // Online softmax: row max / exp / row sum (reduce across 16-lane halves).
    float mnew[8], alpha[8];
#pragma unroll
    for (int r = 0; r < 8; ++r) {
      float rm = S[0][r];
#pragma unroll
      for (int nt = 1; nt < 4; ++nt) rm = fmaxf(rm, S[nt][r]);
#pragma unroll
      for (int sh = 1; sh < 16; sh <<= 1) rm = fmaxf(rm, __shfl_xor(rm, sh, 32));
      mnew[r] = fmaxf(mrow[r], rm);
      alpha[r] = exp2f((mrow[r] - mnew[r]) * LOG2E);
      mrow[r] = mnew[r];
    }
#pragma unroll
    for (int r = 0; r < 8; ++r) {
      float rs = 0.0f;
#pragma unroll
      for (int nt = 0; nt < 4; ++nt) {
        const float p = exp2f((S[nt][r] - mnew[r]) * LOG2E);
        rs += p;
        sP[wave][(r + hi * 8) * 64 + nt * 16 + nlo] = (bf16)p;  // C->A via LDS
      }
#pragma unroll
      for (int sh = 1; sh < 16; sh <<= 1) rs += __shfl_xor(rs, sh, 32);
      lrow[r] = lrow[r] * alpha[r] + rs;
#pragma unroll
      for (int ot = 0; ot < 4; ++ot) Oacc[ot][r] = Oacc[ot][r] * alpha[r];
    }
    // Cross-lane LDS dependency within the wave: wait on DS counter.
    wait_ds0();

    // O += P @ V.  B[k=key][n=d] = sVt[d*64 + key] (pairs contiguous).
#pragma unroll
    for (int ks = 0; ks < 2; ++ks) {
      v16bf pf = load_a_frag(sP[wave], 64, lane, ks * 32);
#pragma unroll
      for (int ot = 0; ot < 4; ++ot) {
        v16bf vfrag = load_bT_frag(sVt, 64, lane, ot * 16, ks * 32);
        Oacc[ot] = wmma_bf16(pf, vfrag, Oacc[ot]);
      }
    }
    __syncthreads();
  }

  // Normalize and store y (bf16 feed for the projection GEMM).
#pragma unroll
  for (int r = 0; r < 8; ++r) {
    const float inv = 1.0f / lrow[r];
    const int t = q0 + wave * 16 + r + hi * 8;
    bf16* yrow = Y + (size_t)(b * T + t) * C + h * D;
#pragma unroll
    for (int ot = 0; ot < 4; ++ot)
      yrow[ot * 16 + nlo] = (bf16)(Oacc[ot][r] * inv);
  }
}

// ---------------------------------------------------------------------------
// Host-side launch
// ---------------------------------------------------------------------------
extern "C" void kernel_launch(void* const* d_in, const int* in_sizes, int n_in,
                              void* d_out, int out_size, void* d_ws, size_t ws_size,
                              hipStream_t stream) {
  (void)in_sizes; (void)n_in; (void)out_size; (void)ws_size;
  const float* x      = (const float*)d_in[0];
  const float* W_qkv  = (const float*)d_in[1];
  const float* b_qkv  = (const float*)d_in[2];
  const float* W_proj = (const float*)d_in[3];
  const float* b_proj = (const float*)d_in[4];
  float* out = (float*)d_out;

  const int Bn = 4, T = 2048, C = 1024, H = 16;
  const int M = Bn * T;  // 8192

  char* ws = (char*)d_ws;
  const size_t MB = 1024u * 1024u;
  bf16* xb   = (bf16*)(ws + 0 * MB);    // 8192x1024    bf16 = 16 MB
  bf16* wqb  = (bf16*)(ws + 16 * MB);   // 1024x3072    bf16 =  6 MB
  bf16* wpb  = (bf16*)(ws + 22 * MB);   // 1024x1024    bf16 =  2 MB
  bf16* qkvb = (bf16*)(ws + 24 * MB);   // 8192x3072    bf16 = 48 MB
  bf16* yb   = (bf16*)(ws + 72 * MB);   // 8192x1024    bf16 = 16 MB

  const int nx = M * C, nwq = C * 3 * C, nwp = C * C;
  cvt_f32_bf16_kernel<<<(nx / 4 + 255) / 256, 256, 0, stream>>>(x, xb, nx);
  cvt_f32_bf16_kernel<<<(nwq / 4 + 255) / 256, 256, 0, stream>>>(W_qkv, wqb, nwq);
  cvt_f32_bf16_kernel<<<(nwp / 4 + 255) / 256, 256, 0, stream>>>(W_proj, wpb, nwp);

  dim3 g1(3 * C / 64, M / 128);
  gemm_bf16_kernel<true><<<g1, 256, 0, stream>>>(xb, wqb, b_qkv, (void*)qkvb,
                                                 M, 3 * C, C);

  dim3 g2(T / 128, H, Bn);
  flash_attn_kernel<<<g2, 256, 0, stream>>>(qkvb, yb, Bn, T, C, H);

  dim3 g3(C / 64, M / 128);
  gemm_bf16_kernel<false><<<g3, 256, 0, stream>>>(yb, wpb, b_proj, (void*)out,
                                                  M, C, C);
}
